// Decoder_22943715295650
// MI455X (gfx1250) — matrix operand order repaired and checked by reading, not compiled, not run on hardware
//
#include <hip/hip_runtime.h>
#include <hip/hip_bf16.h>
#include <cstdint>

typedef __attribute__((ext_vector_type(16))) _Float16 v16h;
typedef __attribute__((ext_vector_type(8)))  _Float16 v8h;
typedef __attribute__((ext_vector_type(8)))  float    v8f;

#define T_DEC 64
#define BATCH 16
#define HID   1024
#define VOCAB 32000
#define WIN   50
#define WL    101
#define LSTM_BLOCKS 32

// ---------------------------------------------------------------------------
// helpers
// ---------------------------------------------------------------------------
__device__ __forceinline__ float sigm(float x) { return 1.0f / (1.0f + expf(-x)); }

// A fragment for v_wmma_f32_16x16x32_f16:
// lane<16 : h[0..15]=K kb+0..15   (row = lane&15)
// lane>=16: h[0..15]=K kb+16..31
__device__ __forceinline__ v16h load_a_frag(const _Float16* __restrict__ arow, int kb, int hi) {
  v8h lo = *(const v8h*)(arow + kb + (hi << 4));
  v8h hv = *(const v8h*)(arow + kb + (hi << 4) + 8);
  return __builtin_shufflevector(lo, hv, 0,1,2,3,4,5,6,7,8,9,10,11,12,13,14,15);
}

__device__ __forceinline__ v8f wmma_f16(v16h a, v16h b, v8f c) {
  return __builtin_amdgcn_wmma_f32_16x16x32_f16(false, a, false, b, (short)0, c, false, false);
}

// D-tile store: VGPR j -> M = j + 8*(lane>=16), N = lane&15
__device__ __forceinline__ void store_tile(v8f acc, float* __restrict__ out,
                                           _Float16* __restrict__ out16,
                                           int m0, int n, int N, float bs,
                                           int relu, int hi, int l15) {
#pragma unroll
  for (int j = 0; j < 8; ++j) {
    int m = m0 + j + (hi << 3);
    float v = acc[j] + bs;
    if (relu) v = v > 0.0f ? v : 0.0f;
    size_t idx = (size_t)m * N + n + l15;
    out[idx] = v;
    if (out16) out16[idx] = (_Float16)v;
  }
}

// grid-wide barrier (bar[0]=count, bar[1]=generation); init to zero by memset
__device__ __forceinline__ void grid_barrier(unsigned* bar, int nblocks) {
  __syncthreads();
  if (threadIdx.x == 0) {
    __threadfence();
    volatile unsigned* gen = bar + 1;
    unsigned g = *gen;
    if (atomicAdd(bar, 1u) == (unsigned)(nblocks - 1)) {
      atomicExch(bar, 0u);
      __threadfence();
      atomicAdd((unsigned*)(bar + 1), 1u);
    } else {
      while (*gen == g) { __builtin_amdgcn_s_sleep(2); }
    }
  }
  __syncthreads();
}

// ---------------------------------------------------------------------------
// fp32 -> f16 conversion
// ---------------------------------------------------------------------------
__global__ __launch_bounds__(256) void cvt_f32_f16(const float* __restrict__ src,
                                                   _Float16* __restrict__ dst, size_t n) {
  size_t i = (size_t)blockIdx.x * 256 + threadIdx.x;
  size_t stride = (size_t)gridDim.x * 256;
  for (; i < n; i += stride) dst[i] = (_Float16)src[i];
}

// embedding gather -> f16 activations, layout (T*B, H)
__global__ __launch_bounds__(256) void embed_kernel(const int* __restrict__ words,
                                                    const float* __restrict__ emb,
                                                    _Float16* __restrict__ x16) {
  size_t i = (size_t)blockIdx.x * 256 + threadIdx.x;
  if (i >= (size_t)T_DEC * BATCH * HID) return;
  int row = (int)(i >> 10);
  int col = (int)(i & 1023);
  x16[i] = (_Float16)emb[(size_t)words[row] * HID + col];
}

// ---------------------------------------------------------------------------
// WMMA GEMM, 2x2 register tiling per wave:
//   OUT[M][N] = A16[M][K] @ W16[N][K]^T + bias0 + bias1
// Each wave owns a 32x32 output macro-tile (2 M-tiles x 2 N-tiles) so A/B
// fragments are each reused twice -> 2 VMEM issues per WMMA instead of 4.
// grid = (N/256, M/32), block = 256 (8 waves).
// ---------------------------------------------------------------------------
__global__ __launch_bounds__(256) void gemm_f16_2x2_kernel(
    const _Float16* __restrict__ A, const _Float16* __restrict__ Wt,
    float* __restrict__ out, _Float16* __restrict__ out16,
    const float* __restrict__ bias0, const float* __restrict__ bias1,
    int N, int K, int relu) {
  const int lane = threadIdx.x & 31;
  const int wave = threadIdx.x >> 5;
  const int hi   = lane >> 4;
  const int l15  = lane & 15;
  const int m0 = blockIdx.y << 5;                   // 32 rows per block
  const int n0 = (blockIdx.x << 8) + (wave << 5);   // 256 cols per block

  const _Float16* arow0 = A + (size_t)(m0 + l15) * K;
  const _Float16* arow1 = arow0 + (size_t)16 * K;
  const _Float16* brow0 = Wt + (size_t)(n0 + l15) * K + (hi << 4);
  const _Float16* brow1 = brow0 + (size_t)16 * K;

  v8f acc00 = {}, acc01 = {}, acc10 = {}, acc11 = {};
  for (int kb = 0; kb < K; kb += 32) {
    v16h a0 = load_a_frag(arow0, kb, hi);
    v16h a1 = load_a_frag(arow1, kb, hi);
    v16h b0 = *(const v16h*)(brow0 + kb);
    v16h b1 = *(const v16h*)(brow1 + kb);
    acc00 = wmma_f16(a0, b0, acc00);
    acc01 = wmma_f16(a0, b1, acc01);
    acc10 = wmma_f16(a1, b0, acc10);
    acc11 = wmma_f16(a1, b1, acc11);
  }
  float bs0 = 0.0f, bs1 = 0.0f;
  if (bias0) { bs0 += bias0[n0 + l15]; bs1 += bias0[n0 + 16 + l15]; }
  if (bias1) { bs0 += bias1[n0 + l15]; bs1 += bias1[n0 + 16 + l15]; }
  store_tile(acc00, out, out16, m0,      n0,      N, bs0, relu, hi, l15);
  store_tile(acc01, out, out16, m0,      n0 + 16, N, bs1, relu, hi, l15);
  store_tile(acc10, out, out16, m0 + 16, n0,      N, bs0, relu, hi, l15);
  store_tile(acc11, out, out16, m0 + 16, n0 + 16, N, bs1, relu, hi, l15);
}

// ---------------------------------------------------------------------------
// Persistent LSTM layer: 32 blocks, internal loop over 64 timesteps.
// Block nb owns h-columns [nb*32, nb*32+32); computes all four gate chunks
// for those columns (waves 0..7 -> gate = wave>>1, sub-tile = wave&1).
// ---------------------------------------------------------------------------
__global__ __launch_bounds__(256) void lstm_layer_kernel(
    const float* __restrict__ Gpre,       // (T*B, 4H) = x@Wih^T + bih + bhh
    const _Float16* __restrict__ Whh16,   // (4H, H) f16
    const float* __restrict__ h0,         // (B,H)
    const float* __restrict__ c0,         // (B,H)
    _Float16* __restrict__ hping,         // 2 * B*H halves (ping-pong)
    _Float16* __restrict__ yf16,          // f16 output, row stride ys, col off yo
    int ys, int yo,
    float* __restrict__ y32,              // optional fp32 output (T*B,H)
    float* __restrict__ hT, float* __restrict__ cT,  // (B,H) finals
    unsigned* __restrict__ bar) {
  __shared__ float gt[16][128];
  const int tid  = threadIdx.x;
  const int lane = tid & 31, wave = tid >> 5;
  const int hi = lane >> 4, l15 = lane & 15;
  const int nb = blockIdx.x;
  const int gate = wave >> 1;
  const int ncol = (gate << 10) + (nb << 5) + ((wave & 1) << 4);  // abs col in 4H
  const _Float16* brow = Whh16 + (size_t)(ncol + l15) * HID + (hi << 4);

  // init h ping buffer (disjoint slices across blocks)
  for (int e = tid; e < 512; e += 256) {
    int idx = (nb << 9) + e;
    hping[idx] = (_Float16)h0[idx];
  }
  // this thread's two (b, hcol) update elements (fixed across time)
  const int e0 = tid, e1 = tid + 256;
  const int b0_ = e0 >> 5, col0 = e0 & 31;
  const int b1_ = e1 >> 5, col1 = e1 & 31;
  const int hc0 = (nb << 5) + col0, hc1 = (nb << 5) + col1;
  float cr0 = c0[(b0_ << 10) + hc0];
  float cr1 = c0[(b1_ << 10) + hc1];
  float hr0 = 0.0f, hr1 = 0.0f;

  grid_barrier(bar, LSTM_BLOCKS);

  for (int t = 0; t < T_DEC; ++t) {
    const _Float16* hb = hping + ((t & 1) << 14);  // B*H = 16384
    const _Float16* arow = hb + (l15 << 10);
    v8f acc = {};
#pragma unroll 4
    for (int kb = 0; kb < HID; kb += 32) {
      v16h af = load_a_frag(arow, kb, hi);
      v16h bf = *(const v16h*)(brow + kb);
      acc = wmma_f16(af, bf, acc);
    }
#pragma unroll
    for (int j = 0; j < 8; ++j)
      gt[j + (hi << 3)][(wave << 4) + l15] = acc[j];
    __syncthreads();

    const float* gp = Gpre + (size_t)(t * BATCH) * 4096;
    {
      float gi = gt[b0_][col0]      + gp[(b0_ << 12) + hc0];
      float gf = gt[b0_][col0 + 32] + gp[(b0_ << 12) + 1024 + hc0];
      float gg = gt[b0_][col0 + 64] + gp[(b0_ << 12) + 2048 + hc0];
      float go = gt[b0_][col0 + 96] + gp[(b0_ << 12) + 3072 + hc0];
      cr0 = sigm(gf) * cr0 + sigm(gi) * tanhf(gg);
      hr0 = sigm(go) * tanhf(cr0);
      hping[(((t + 1) & 1) << 14) + (b0_ << 10) + hc0] = (_Float16)hr0;
      yf16[(size_t)(t * BATCH + b0_) * ys + yo + hc0] = (_Float16)hr0;
      if (y32) y32[(size_t)(t * BATCH + b0_) * HID + hc0] = hr0;
    }
    {
      float gi = gt[b1_][col1]      + gp[(b1_ << 12) + hc1];
      float gf = gt[b1_][col1 + 32] + gp[(b1_ << 12) + 1024 + hc1];
      float gg = gt[b1_][col1 + 64] + gp[(b1_ << 12) + 2048 + hc1];
      float go = gt[b1_][col1 + 96] + gp[(b1_ << 12) + 3072 + hc1];
      cr1 = sigm(gf) * cr1 + sigm(gi) * tanhf(gg);
      hr1 = sigm(go) * tanhf(cr1);
      hping[(((t + 1) & 1) << 14) + (b1_ << 10) + hc1] = (_Float16)hr1;
      yf16[(size_t)(t * BATCH + b1_) * ys + yo + hc1] = (_Float16)hr1;
      if (y32) y32[(size_t)(t * BATCH + b1_) * HID + hc1] = hr1;
    }
    grid_barrier(bar, LSTM_BLOCKS);
  }
  hT[(b0_ << 10) + hc0] = hr0;
  hT[(b1_ << 10) + hc1] = hr1;
  cT[(b0_ << 10) + hc0] = cr0;
  cT[(b1_ << 10) + hc1] = cr1;
}

// ---------------------------------------------------------------------------
// attention position predictor: p, wstart per (b,t).  grid = B*T blocks.
// ---------------------------------------------------------------------------
__global__ __launch_bounds__(256) void attn_p_kernel(
    const float* __restrict__ y1, const float* __restrict__ aW1,
    const float* __restrict__ ab1, const float* __restrict__ aW2,
    const float* __restrict__ ab2, const int* __restrict__ lengths,
    float* __restrict__ p_arr, int* __restrict__ ws_arr) {
  __shared__ float ht[HID];
  __shared__ float red[8];
  const int bid = blockIdx.x;           // b*64 + t
  const int b = bid >> 6, t = bid & 63;
  const int lane = threadIdx.x & 31, wave = threadIdx.x >> 5;
  const float* hrow = y1 + (size_t)(t * BATCH + b) * HID;
  for (int k = threadIdx.x; k < HID; k += 256) ht[k] = hrow[k];
  __syncthreads();
  float wacc = 0.0f;
  for (int r = wave; r < 512; r += 8) {
    const float* wrow = aW1 + (size_t)r * HID;
    float s = 0.0f;
    for (int k = lane; k < HID; k += 32) s += wrow[k] * ht[k];
#pragma unroll
    for (int off = 16; off; off >>= 1) s += __shfl_xor(s, off, 32);
    if (lane == 0) wacc += tanhf(s + ab1[r]) * aW2[r];
  }
  if (lane == 0) red[wave] = wacc;
  __syncthreads();
  if (threadIdx.x == 0) {
    float tot = 0.0f;
#pragma unroll
    for (int i = 0; i < 8; ++i) tot += red[i];
    float pv = sigm(tot + ab2[0]);
    float p = (float)WIN + (float)lengths[b] * pv;
    p_arr[bid] = p;
    ws_arr[bid] = (int)rintf(p - (float)WIN);
  }
}

// ---------------------------------------------------------------------------
// attention scores + softmax*gauss + context.  grid = B*T blocks.
// ---------------------------------------------------------------------------
__global__ __launch_bounds__(256) void attn_score_ctx_kernel(
    const float* __restrict__ enc, const float* __restrict__ y1,
    const float* __restrict__ p_arr, const int* __restrict__ ws_arr,
    const int* __restrict__ lengths,
    float* __restrict__ a_out, _Float16* __restrict__ cat) {
  __shared__ float ht[HID];
  __shared__ float sw[WL];
  __shared__ float aw[WL];
  const int bid = blockIdx.x;
  const int b = bid >> 6, t = bid & 63;
  const int lane = threadIdx.x & 31, wave = threadIdx.x >> 5;
  const float* hrow = y1 + (size_t)(t * BATCH + b) * HID;
  for (int k = threadIdx.x; k < HID; k += 256) ht[k] = hrow[k];
  const int wstart = ws_arr[bid];
  const float p = p_arr[bid];
  const int len = lengths[b];
  __syncthreads();
  for (int w = wave; w < WL; w += 8) {
    const int pos = wstart + w;
    const float* e = enc + ((size_t)pos * BATCH + b) * HID;
    float s = 0.0f;
    for (int k = lane; k < HID; k += 32) s += ht[k] * e[k];
#pragma unroll
    for (int off = 16; off; off >>= 1) s += __shfl_xor(s, off, 32);
    if (lane == 0) {
      bool valid = (pos >= WIN) && (pos < len + WIN);
      sw[w] = valid ? s : 1e-14f;
    }
  }
  __syncthreads();
  if (threadIdx.x == 0) {
    float mx = -1e30f;
    for (int w = 0; w < WL; ++w) mx = fmaxf(mx, sw[w]);
    float sum = 0.0f;
    for (int w = 0; w < WL; ++w) { float e = expf(sw[w] - mx); aw[w] = e; sum += e; }
    float inv = 1.0f / sum;
    for (int w = 0; w < WL; ++w) {
      float d = (float)(wstart + w) - p;
      float g = expf(-d * d * (1.0f / 1250.0f));   // 2*STD_SQ = 1250
      float av = aw[w] * inv * g;
      aw[w] = av;
      a_out[(size_t)bid * WL + w] = av;
    }
  }
  __syncthreads();
  for (int h = threadIdx.x; h < HID; h += 256) {
    float s = 0.0f;
    for (int w = 0; w < WL; ++w)
      s += aw[w] * enc[((size_t)(wstart + w) * BATCH + b) * HID + h];
    cat[(size_t)(t * BATCH + b) * 2048 + h] = (_Float16)s;
  }
}

// ---------------------------------------------------------------------------
// final attention-weights scatter for b==0.  grid = T blocks x 128 threads.
// ---------------------------------------------------------------------------
__global__ void weights_kernel(const float* __restrict__ a_buf,
                               const int* __restrict__ ws_arr,
                               float* __restrict__ wout, int s0) {
  const int t = blockIdx.x, w = threadIdx.x;
  if (w < WL) {
    int src = ws_arr[t] + w - WIN;   // bid for b==0 is t
    if (src >= 0 && src < s0)
      atomicAdd(&wout[(size_t)t * s0 + src], a_buf[(size_t)t * WL + w]);
  }
}

// ---------------------------------------------------------------------------
// host
// ---------------------------------------------------------------------------
extern "C" void kernel_launch(void* const* d_in, const int* in_sizes, int n_in,
                              void* d_out, int out_size, void* d_ws, size_t ws_size,
                              hipStream_t stream) {
  const float* enc     = (const float*)d_in[0];
  const float* h0      = (const float*)d_in[1];
  const float* c0      = (const float*)d_in[2];
  const int*   words   = (const int*)d_in[3];
  const int*   lengths = (const int*)d_in[4];
  const float* emb     = (const float*)d_in[5];
  const float* Wih0    = (const float*)d_in[6];
  const float* Whh0    = (const float*)d_in[7];
  const float* bih0    = (const float*)d_in[8];
  const float* bhh0    = (const float*)d_in[9];
  const float* Wih1    = (const float*)d_in[10];
  const float* Whh1    = (const float*)d_in[11];
  const float* bih1    = (const float*)d_in[12];
  const float* bhh1    = (const float*)d_in[13];
  const float* aW1     = (const float*)d_in[14];
  const float* ab1     = (const float*)d_in[15];
  const float* aW2     = (const float*)d_in[16];
  const float* ab2     = (const float*)d_in[17];
  const float* dW1     = (const float*)d_in[18];
  const float* db1     = (const float*)d_in[19];
  const float* dW2     = (const float*)d_in[20];
  const float* db2     = (const float*)d_in[21];

  float* out = (float*)d_out;
  const size_t Y_N    = (size_t)T_DEC * BATCH * VOCAB;   // 32,768,000
  const size_t MID_N  = (size_t)T_DEC * BATCH * HID;     //  1,048,576
  const size_t MID_OFF = Y_N;
  const size_t HT_OFF  = MID_OFF + MID_N;
  const size_t CT_OFF  = HT_OFF + (size_t)2 * BATCH * HID;
  const size_t W_OFF   = CT_OFF + (size_t)2 * BATCH * HID;
  const int s0 = (int)(((size_t)out_size - W_OFF) / T_DEC);

  // workspace carve-up
  char* ws = (char*)d_ws;
  size_t off = 0;
  auto carve = [&](size_t bytes) -> char* {
    char* p = ws + off;
    off = (off + bytes + 255) & ~(size_t)255;
    return p;
  };
  unsigned* bar   = (unsigned*)carve(256);
  _Float16* xf16  = (_Float16*)carve((size_t)T_DEC * BATCH * HID * 2);
  _Float16* Wih0h = (_Float16*)carve((size_t)4096 * HID * 2);
  _Float16* Whh0h = (_Float16*)carve((size_t)4096 * HID * 2);
  _Float16* Wih1h = (_Float16*)carve((size_t)4096 * HID * 2);
  _Float16* Whh1h = (_Float16*)carve((size_t)4096 * HID * 2);
  _Float16* dW1h  = (_Float16*)carve((size_t)HID * 2048 * 2);
  _Float16* dW2h  = (_Float16*)carve((size_t)VOCAB * HID * 2);
  float*    G     = (float*)carve((size_t)T_DEC * BATCH * 4096 * 4);
  _Float16* y0h   = (_Float16*)carve((size_t)T_DEC * BATCH * HID * 2);
  float*    y1f   = (float*)carve((size_t)T_DEC * BATCH * HID * 4);
  _Float16* cat   = (_Float16*)carve((size_t)T_DEC * BATCH * 2048 * 2);
  _Float16* midh  = (_Float16*)carve((size_t)T_DEC * BATCH * HID * 2);
  _Float16* hping = (_Float16*)carve((size_t)2 * BATCH * HID * 2);
  float*    p_arr = (float*)carve((size_t)BATCH * T_DEC * 4);
  int*      wsarr = (int*)carve((size_t)BATCH * T_DEC * 4);
  float*    a_buf = (float*)carve((size_t)BATCH * T_DEC * WL * 4);

  hipMemsetAsync(bar, 0, 256, stream);
  hipMemsetAsync(out + W_OFF, 0, (size_t)T_DEC * s0 * 4, stream);

  // f16 weight staging
  cvt_f32_f16<<<4096, 256, 0, stream>>>(Wih0, Wih0h, (size_t)4096 * HID);
  cvt_f32_f16<<<4096, 256, 0, stream>>>(Whh0, Whh0h, (size_t)4096 * HID);
  cvt_f32_f16<<<4096, 256, 0, stream>>>(Wih1, Wih1h, (size_t)4096 * HID);
  cvt_f32_f16<<<4096, 256, 0, stream>>>(Whh1, Whh1h, (size_t)4096 * HID);
  cvt_f32_f16<<<4096, 256, 0, stream>>>(dW1, dW1h, (size_t)HID * 2048);
  cvt_f32_f16<<<8192, 256, 0, stream>>>(dW2, dW2h, (size_t)VOCAB * HID);

  // embedding gather
  embed_kernel<<<(T_DEC * BATCH * HID + 255) / 256, 256, 0, stream>>>(words, emb, xf16);

  // layer 0: batched input GEMM then recurrent scan
  gemm_f16_2x2_kernel<<<dim3(4096 / 256, T_DEC * BATCH / 32), 256, 0, stream>>>(
      xf16, Wih0h, G, nullptr, bih0, bhh0, 4096, HID, 0);
  lstm_layer_kernel<<<LSTM_BLOCKS, 256, 0, stream>>>(
      G, Whh0h, h0, c0, hping, y0h, HID, 0, nullptr,
      out + HT_OFF, out + CT_OFF, bar);

  // layer 1
  gemm_f16_2x2_kernel<<<dim3(4096 / 256, T_DEC * BATCH / 32), 256, 0, stream>>>(
      y0h, Wih1h, G, nullptr, bih1, bhh1, 4096, HID, 0);
  lstm_layer_kernel<<<LSTM_BLOCKS, 256, 0, stream>>>(
      G, Whh1h, h0 + (size_t)BATCH * HID, c0 + (size_t)BATCH * HID,
      hping, cat, 2048, 1024, y1f,
      out + HT_OFF + (size_t)BATCH * HID, out + CT_OFF + (size_t)BATCH * HID, bar);

  // attention
  attn_p_kernel<<<BATCH * T_DEC, 256, 0, stream>>>(y1f, aW1, ab1, aW2, ab2,
                                                   lengths, p_arr, wsarr);
  attn_score_ctx_kernel<<<BATCH * T_DEC, 256, 0, stream>>>(enc, y1f, p_arr, wsarr,
                                                           lengths, a_buf, cat);

  // mid = relu([ctx,y1] @ dW1^T + db1)  -> fp32 out + f16 staging for vocab GEMM
  gemm_f16_2x2_kernel<<<dim3(HID / 256, T_DEC * BATCH / 32), 256, 0, stream>>>(
      cat, dW1h, out + MID_OFF, midh, db1, nullptr, HID, 2048, 1);

  // y = mid @ dW2^T + db2   (dominant GEMM: 67 GFLOP, ~196 MB HBM traffic)
  gemm_f16_2x2_kernel<<<dim3(VOCAB / 256, T_DEC * BATCH / 32), 256, 0, stream>>>(
      midh, dW2h, out, nullptr, db2, nullptr, VOCAB, HID, 0);

  // attention weights for b==0
  weights_kernel<<<T_DEC, 128, 0, stream>>>(a_buf, wsarr, out + W_OFF, s0);
}
